// Qwen2MLP_50474455662978
// MI455X (gfx1250) — compile-verified, matrix operand order checked by reading, not run on hardware
//
#include <hip/hip_runtime.h>
#include <hip/hip_bf16.h>

// ---------------------------------------------------------------------------
// Qwen2 MLP, int8-coded weights, MI455X (gfx1250, wave32, WMMA bf16).
// Pre-passes dequantize weights / convert x to bf16 once (bandwidth-bound,
// ~3 us at 23.3 TB/s) so the two GEMM kernels are pure WMMA loops with
// copy-only async global->LDS staging (ASYNCcnt) and double-buffered LDS.
// LDS tiles use a 72-element (144 B = 36 bank) row stride: gcd(36,64)=4 puts
// the 16 fragment rows on disjoint 4-bank groups -> conflict-free ds_load_b128.
// ---------------------------------------------------------------------------

#define DIM 2048
#define HID 5632

#define TBM 128      // block tile M
#define TBN 128      // block tile N
#define TBK 64       // staged K per tile (2 WMMA K-steps)
#define LDST 72      // padded LDS row stride (elements): 144 B, conflict-free

typedef __attribute__((ext_vector_type(16))) __bf16 v16bf;
typedef __attribute__((ext_vector_type(8)))  float  v8f;

struct Frag { union { v16bf v; uint4 q[2]; }; };

// ---- packed fp32 -> bf16 ---------------------------------------------------
#if __has_builtin(__builtin_amdgcn_cvt_pk_bf16_f32)
typedef __attribute__((ext_vector_type(2))) __bf16 v2bf;
__device__ __forceinline__ unsigned int pk2(float lo, float hi) {
  v2bf r = __builtin_amdgcn_cvt_pk_bf16_f32(lo, hi);
  return __builtin_bit_cast(unsigned int, r);
}
#else
__device__ __forceinline__ unsigned short f2bf(float f) {
  unsigned int u = __builtin_bit_cast(unsigned int, f);
  u += 0x7FFFu + ((u >> 16) & 1u);
  return (unsigned short)(u >> 16);
}
__device__ __forceinline__ unsigned int pk2(float lo, float hi) {
  return (unsigned int)f2bf(lo) | ((unsigned int)f2bf(hi) << 16);
}
#endif

__device__ __forceinline__ float silu_f(float g) {
  float e = __builtin_amdgcn_exp2f(-1.44269504f * g);
  return g * __builtin_amdgcn_rcpf(1.0f + e);
}

// ---- global -> LDS staging (async path when available) ---------------------
#if __has_builtin(__builtin_amdgcn_global_load_async_to_lds_b128) && \
    __has_builtin(__builtin_amdgcn_s_wait_asynccnt)
#define HAVE_ASYNC 1
typedef __attribute__((ext_vector_type(4))) int v4i;
typedef __attribute__((address_space(1))) v4i* as1_v4i;
typedef __attribute__((address_space(3))) v4i* as3_v4i;
#else
#define HAVE_ASYNC 0
#endif

__device__ __forceinline__ void stage16(const unsigned short* g, unsigned short* l) {
#if HAVE_ASYNC
  unsigned short* gnc = const_cast<unsigned short*>(g);
  __builtin_amdgcn_global_load_async_to_lds_b128((as1_v4i)gnc, (as3_v4i)l, 0, 0);
#else
  *(uint4*)l = *(const uint4*)g;
#endif
}
__device__ __forceinline__ void stage_wait() {
#if HAVE_ASYNC
  __builtin_amdgcn_s_wait_asynccnt(0);
#endif
}

// Load one 16x32 bf16 fragment from an LDS tile with padded row stride LDST.
__device__ __forceinline__ void load_frag(Frag& f, const unsigned short* tile,
                                          int row, int koff, int half) {
  const unsigned short* p = tile + row * LDST + koff;
  f.q[0] = *(const uint4*)(p + half * 8);
  f.q[1] = *(const uint4*)(p + 16 + half * 8);
}

// ---------------------------------------------------------------------------
// Pre-pass 1: x fp32 -> bf16
// ---------------------------------------------------------------------------
__global__ __launch_bounds__(256)
void cvtx_kernel(const float* __restrict__ x, unsigned short* __restrict__ xb) {
  const size_t e0 = ((size_t)blockIdx.x * 256 + threadIdx.x) * 8;
  float4 a0 = ((const float4*)(x + e0))[0];
  float4 a1 = ((const float4*)(x + e0))[1];
  uint4 o;
  o.x = pk2(a0.x, a0.y); o.y = pk2(a0.z, a0.w);
  o.z = pk2(a1.x, a1.y); o.w = pk2(a1.z, a1.w);
  *(uint4*)(xb + e0) = o;
}

// ---------------------------------------------------------------------------
// Pre-pass 2: int32 codes -> bf16, w = code*s - 128*s (one FMA per element)
// ---------------------------------------------------------------------------
__global__ __launch_bounds__(256)
void dequant_kernel(const int* __restrict__ wq, const float* __restrict__ scale,
                    unsigned short* __restrict__ wb, int rowlen) {
  const size_t e0 = ((size_t)blockIdx.x * 256 + threadIdx.x) * 8;
  const int row = (int)(e0 / (size_t)rowlen);
  const float s = scale[row];
  const float b = -128.0f * s;
  int4 c0 = ((const int4*)(wq + e0))[0];
  int4 c1 = ((const int4*)(wq + e0))[1];
  uint4 o;
  o.x = pk2(fmaf((float)c0.x, s, b), fmaf((float)c0.y, s, b));
  o.y = pk2(fmaf((float)c0.z, s, b), fmaf((float)c0.w, s, b));
  o.z = pk2(fmaf((float)c1.x, s, b), fmaf((float)c1.y, s, b));
  o.w = pk2(fmaf((float)c1.z, s, b), fmaf((float)c1.w, s, b));
  *(uint4*)(wb + e0) = o;
}

// ---------------------------------------------------------------------------
// Kernel: fused gate+up bf16 GEMM, h = silu(gate)*up -> bf16 [M, HID]
// 128x128x64 block tile, 8 waves of 64x32, double-buffered async LDS.
// ---------------------------------------------------------------------------
__global__ __launch_bounds__(256)
void qwen_gateup_kernel(const unsigned short* __restrict__ xb,
                        const unsigned short* __restrict__ wg,
                        const unsigned short* __restrict__ wu,
                        unsigned short* __restrict__ h) {
  __shared__ __align__(16) unsigned short As[2][TBM * LDST];
  __shared__ __align__(16) unsigned short Bg[2][TBN * LDST];
  __shared__ __align__(16) unsigned short Bu[2][TBN * LDST];

  const int tid  = threadIdx.x;
  const int gm   = blockIdx.y * TBM;
  const int gn   = blockIdx.x * TBN;
  const int lane = tid & 31;
  const int wid  = tid >> 5;
  const int wm   = (wid >> 2) * 64;   // 0 / 64
  const int wn   = (wid & 3) * 32;    // 0..96
  const int frow = lane & 15;
  const int half = lane >> 4;

  // 128x64 tile = 1024 16B chunks; 4 per thread (8 chunks per 64-elem row)
  auto stage = [&](int k0, int buf) {
#pragma unroll
    for (int i = 0; i < 4; ++i) {
      const int ci  = tid + 256 * i;
      const int row = ci >> 3;
      const int col = (ci & 7) * 8;
      const int lo  = row * LDST + col;
      stage16(xb + (size_t)(gm + row) * DIM + k0 + col, &As[buf][lo]);
      stage16(wg + (size_t)(gn + row) * DIM + k0 + col, &Bg[buf][lo]);
      stage16(wu + (size_t)(gn + row) * DIM + k0 + col, &Bu[buf][lo]);
    }
  };

  v8f accg[4][2] = {};
  v8f accu[4][2] = {};

  stage(0, 0);
  stage_wait();
  __syncthreads();

  const int nk = DIM / TBK;   // 32
  for (int kt = 0; kt < nk; ++kt) {
    const int buf = kt & 1;
    if (kt + 1 < nk) stage((kt + 1) * TBK, buf ^ 1);

#pragma unroll
    for (int ks = 0; ks < 2; ++ks) {
      const int koff = ks * 32;
      Frag af[4], bg[2], bu[2];
#pragma unroll
      for (int i = 0; i < 4; ++i)
        load_frag(af[i], As[buf], wm + i * 16 + frow, koff, half);
#pragma unroll
      for (int j = 0; j < 2; ++j) {
        load_frag(bg[j], Bg[buf], wn + j * 16 + frow, koff, half);
        load_frag(bu[j], Bu[buf], wn + j * 16 + frow, koff, half);
      }
#pragma unroll
      for (int i = 0; i < 4; ++i)
#pragma unroll
        for (int j = 0; j < 2; ++j) {
          accg[i][j] = __builtin_amdgcn_wmma_f32_16x16x32_bf16(
              false, af[i].v, false, bg[j].v, (short)0, accg[i][j], false, false);
          accu[i][j] = __builtin_amdgcn_wmma_f32_16x16x32_bf16(
              false, af[i].v, false, bu[j].v, (short)0, accu[i][j], false, false);
        }
    }
    stage_wait();
    __syncthreads();
  }

  // epilogue: h = silu(gate)*up -> bf16
  const int rbase = half << 3;        // 0 or 8
  const int cn    = frow;
#pragma unroll
  for (int i = 0; i < 4; ++i)
#pragma unroll
    for (int j = 0; j < 2; ++j) {
      size_t row0 = (size_t)(gm + wm + i * 16 + rbase);
      int    col  = gn + wn + j * 16 + cn;
#pragma unroll
      for (int r = 0; r < 8; ++r) {
        float g = accg[i][j][r];
        float u = accu[i][j][r];
        unsigned int p = pk2(silu_f(g) * u, 0.0f);
        h[(row0 + r) * HID + col] = (unsigned short)p;
      }
    }
}

// ---------------------------------------------------------------------------
// Kernel: out = h @ Wd^T  (all bf16 in, fp32 out)
// ---------------------------------------------------------------------------
__global__ __launch_bounds__(256)
void qwen_down_kernel(const unsigned short* __restrict__ h,
                      const unsigned short* __restrict__ wd,
                      float* __restrict__ out) {
  __shared__ __align__(16) unsigned short As[2][TBM * LDST];
  __shared__ __align__(16) unsigned short Bs[2][TBN * LDST];

  const int tid  = threadIdx.x;
  const int gm   = blockIdx.y * TBM;
  const int gn   = blockIdx.x * TBN;
  const int lane = tid & 31;
  const int wid  = tid >> 5;
  const int wm   = (wid >> 2) * 64;
  const int wn   = (wid & 3) * 32;
  const int frow = lane & 15;
  const int half = lane >> 4;

  auto stage = [&](int k0, int buf) {
#pragma unroll
    for (int i = 0; i < 4; ++i) {
      const int ci  = tid + 256 * i;
      const int row = ci >> 3;
      const int col = (ci & 7) * 8;
      const int lo  = row * LDST + col;
      stage16(h  + (size_t)(gm + row) * HID + k0 + col, &As[buf][lo]);
      stage16(wd + (size_t)(gn + row) * HID + k0 + col, &Bs[buf][lo]);
    }
  };

  v8f acc[4][2] = {};

  stage(0, 0);
  stage_wait();
  __syncthreads();

  const int nk = HID / TBK;   // 88
  for (int kt = 0; kt < nk; ++kt) {
    const int buf = kt & 1;
    if (kt + 1 < nk) stage((kt + 1) * TBK, buf ^ 1);

#pragma unroll
    for (int ks = 0; ks < 2; ++ks) {
      const int koff = ks * 32;
      Frag af[4], bf[2];
#pragma unroll
      for (int i = 0; i < 4; ++i)
        load_frag(af[i], As[buf], wm + i * 16 + frow, koff, half);
#pragma unroll
      for (int j = 0; j < 2; ++j)
        load_frag(bf[j], Bs[buf], wn + j * 16 + frow, koff, half);
#pragma unroll
      for (int i = 0; i < 4; ++i)
#pragma unroll
        for (int j = 0; j < 2; ++j)
          acc[i][j] = __builtin_amdgcn_wmma_f32_16x16x32_bf16(
              false, af[i].v, false, bf[j].v, (short)0, acc[i][j], false, false);
    }
    stage_wait();
    __syncthreads();
  }

  const int rbase = half << 3;
  const int cn    = frow;
#pragma unroll
  for (int i = 0; i < 4; ++i)
#pragma unroll
    for (int j = 0; j < 2; ++j) {
      size_t row0 = (size_t)(gm + wm + i * 16 + rbase);
      int    col  = gn + wn + j * 16 + cn;
#pragma unroll
      for (int r = 0; r < 8; ++r)
        out[(row0 + r) * DIM + col] = acc[i][j][r];
    }
}

// ---------------------------------------------------------------------------
extern "C" void kernel_launch(void* const* d_in, const int* in_sizes, int n_in,
                              void* d_out, int out_size, void* d_ws, size_t ws_size,
                              hipStream_t stream) {
  const float* x   = (const float*)d_in[0];
  const int*   wqg = (const int*)  d_in[1];
  const float* sg  = (const float*)d_in[2];
  const int*   wqu = (const int*)  d_in[3];
  const float* su  = (const float*)d_in[4];
  const int*   wqd = (const int*)  d_in[5];
  const float* sd  = (const float*)d_in[6];
  float* out = (float*)d_out;

  const int Mtot = in_sizes[0] / DIM;            // B*S = 8192

  // workspace layout (bf16): h | x_bf (reused for wd_bf) | wg_bf | wu_bf
  unsigned short* hb  = (unsigned short*)d_ws;
  unsigned short* xb  = hb + (size_t)Mtot * HID;
  unsigned short* wgb = xb + (size_t)Mtot * DIM;
  unsigned short* wub = wgb + (size_t)HID * DIM;
  unsigned short* wdb = xb;                       // reuse after gate/up GEMM

  dim3 blk(256);

  cvtx_kernel<<<dim3((unsigned)((size_t)Mtot * DIM / 8 / 256)), blk, 0, stream>>>(x, xb);
  dequant_kernel<<<dim3((unsigned)((size_t)HID * DIM / 8 / 256)), blk, 0, stream>>>(wqg, sg, wgb, DIM);
  dequant_kernel<<<dim3((unsigned)((size_t)HID * DIM / 8 / 256)), blk, 0, stream>>>(wqu, su, wub, DIM);

  qwen_gateup_kernel<<<dim3(HID / TBN, Mtot / TBM), blk, 0, stream>>>(xb, wgb, wub, hb);

  dequant_kernel<<<dim3((unsigned)((size_t)DIM * HID / 8 / 256)), blk, 0, stream>>>(wqd, sd, wdb, HID);

  qwen_down_kernel<<<dim3(DIM / TBN, Mtot / TBM), blk, 0, stream>>>(hb, wdb, out);
}
